// CIN_944892805619
// MI455X (gfx1250) — matrix operand order repaired and checked by reading, not compiled, not run on hardware
//
#include <hip/hip_runtime.h>
#include <hip/hip_bf16.h>

typedef __attribute__((ext_vector_type(16))) __bf16 v16bf;
typedef __attribute__((ext_vector_type(8)))  float  v8f;
typedef __attribute__((ext_vector_type(4)))  float  v4f;

// ---------------------------------------------------------------------------
// Pre-swizzle W (f32, [K x 128]) into bf16 B-fragment layout:
//   WB[((ks*8 + ntile)*32 + lane)*8 + r] = pack(bf16 W[k,n], bf16 W[k+1,n])
//   k = ks*32 + (lane>=16 ? 16 : 0) + 2r ,  n = ntile*16 + (lane&15)
// This is the contiguous k order (k = 16*half + element) that a_kbase also uses, so the GEMM
// kernels load B with coalesced 32B-per-lane vector loads.
// ---------------------------------------------------------------------------
__global__ __launch_bounds__(256) void prep_w_kernel(
    const float* __restrict__ W, unsigned* __restrict__ WB, int ksteps) {
  int tid = blockIdx.x * 256 + threadIdx.x;
  int total = ksteps << 11;           // ksteps*8*32*8
  if (tid >= total) return;
  int r    = tid & 7;
  int lane = (tid >> 3) & 31;
  int nt   = (tid >> 8) & 7;
  int ks   = tid >> 11;
  int klo  = ks * 32 + ((lane >> 4) << 4) + 2 * r;
  int n    = nt * 16 + (lane & 15);
  float f0 = W[klo * 128 + n];
  float f1 = W[(klo + 1) * 128 + n];
  __bf16 h0 = (__bf16)f0;
  __bf16 h1 = (__bf16)f1;
  unsigned u0 = (unsigned)__builtin_bit_cast(unsigned short, h0);
  unsigned u1 = (unsigned)__builtin_bit_cast(unsigned short, h1);
  WB[tid] = u0 | (u1 << 16);
}

// A-fragment K index for 16-bit A 16x32 layout (matches WB packing):
//   VGPR r : lanes0-15 K=2r,2r+1      lanes16-31 K=16+2r,17+2r
__device__ __forceinline__ int a_kbase(int r, int half) {
  return (half << 4) + (r << 1);
}

// ---------------------------------------------------------------------------
// Layer 0: one WG per batch b.  M rows = d (32), N = 128, K = 1600 (i*40+j).
//   h0[b,s,d] = relu( sum_k x[b,i,d]*x[b,j,d]*W0[k,s] + b0[s] )
//   s <  64 -> store f32 next_hidden to ws
//   s >= 64 -> out[b, s-64] = sum_d h0   (direct0)
// ---------------------------------------------------------------------------
__global__ __launch_bounds__(256) void layer0_kernel(
    const float* __restrict__ x, const unsigned* __restrict__ W0B,
    const float* __restrict__ b0, float* __restrict__ NH,
    float* __restrict__ out) {
  __shared__ float Xt[40 * 33];      // Xt[i*33 + d] = x[b,i,d]
  __shared__ float red[4 * 64];      // [h][half][s-64] partial d-sums

  const int b   = blockIdx.x;
  const int tid = threadIdx.x;
  const float* xb = x + b * 1280;
  for (int idx = tid; idx < 1280; idx += 256) {
    int i = idx >> 5, d = idx & 31;
    Xt[i * 33 + d] = xb[idx];
  }
  __syncthreads();

  const int lane = tid & 31;
  const int wid  = tid >> 5;
  const int h    = wid >> 2;         // M half: d in [16h, 16h+16)
  const int g    = wid & 3;          // N group: tiles 2g, 2g+1
  const int m    = lane & 15;
  const int half = lane >> 4;
  const int d    = h * 16 + m;

  v8f acc0 = {0.f, 0.f, 0.f, 0.f, 0.f, 0.f, 0.f, 0.f};
  v8f acc1 = {0.f, 0.f, 0.f, 0.f, 0.f, 0.f, 0.f, 0.f};

  const unsigned* B0p = W0B + (unsigned)((2 * g) * 32 + lane) * 8;
  const unsigned* B1p = W0B + (unsigned)((2 * g + 1) * 32 + lane) * 8;

  for (int ks = 0; ks < 50; ++ks) {
    v16bf A;
#pragma unroll
    for (int r = 0; r < 8; ++r) {
      int klo = ks * 32 + a_kbase(r, half);
      int khi = klo + 1;
      int i0 = (klo * 205) >> 13;  int j0 = klo - i0 * 40;   // k/40, k%40
      int i1 = (khi * 205) >> 13;  int j1 = khi - i1 * 40;
      float a0 = Xt[i0 * 33 + d] * Xt[j0 * 33 + d];
      float a1 = Xt[i1 * 33 + d] * Xt[j1 * 33 + d];
      A[2 * r]     = (__bf16)a0;
      A[2 * r + 1] = (__bf16)a1;
    }
    const v16bf Bv0 = *(const v16bf*)(B0p + (unsigned)ks * 2048);
    const v16bf Bv1 = *(const v16bf*)(B1p + (unsigned)ks * 2048);
    acc0 = __builtin_amdgcn_wmma_f32_16x16x32_bf16(false, A, false, Bv0,
                                                   (short)0, acc0, false, false);
    acc1 = __builtin_amdgcn_wmma_f32_16x16x32_bf16(false, A, false, Bv1,
                                                   (short)0, acc1, false, false);
  }

  // Epilogue: bias + relu; rows (C layout): VGPR r -> d = 16h + 8*half + r
  const int dofs = h * 16 + half * 8;
#pragma unroll
  for (int t = 0; t < 2; ++t) {
    v8f acc = t ? acc1 : acc0;
    int s = (2 * g + t) * 16 + m;
    float bias = b0[s];
    float v[8];
#pragma unroll
    for (int r = 0; r < 8; ++r) {
      float q = acc[r] + bias;
      v[r] = q > 0.f ? q : 0.f;
    }
    if (g < 2) {                       // s in [0,64): next_hidden
      v4f lo = {v[0], v[1], v[2], v[3]};
      v4f hi = {v[4], v[5], v[6], v[7]};
      float* p = NH + ((unsigned)(b * 64 + s) * 32 + dofs);
      *(v4f*)p = lo;
      *(v4f*)(p + 4) = hi;
    } else {                           // s in [64,128): direct0 d-sum
      float ps = ((v[0] + v[1]) + (v[2] + v[3])) + ((v[4] + v[5]) + (v[6] + v[7]));
      red[(h * 2 + half) * 64 + (s - 64)] = ps;
    }
  }
  __syncthreads();
  if (tid < 64) {
    out[b * 192 + tid] = (red[tid] + red[64 + tid]) + (red[128 + tid] + red[192 + tid]);
  }
}

// ---------------------------------------------------------------------------
// Layer 1: one WG per b.  K = 2560 (i*64+j), features x[b,i,d]*h0[b,j,d].
//   out[b, 64+s] = sum_d relu( gemm + b1[s] )
// ---------------------------------------------------------------------------
__global__ __launch_bounds__(256) void layer1_kernel(
    const float* __restrict__ x, const unsigned* __restrict__ W1B,
    const float* __restrict__ b1, const float* __restrict__ NH,
    float* __restrict__ out) {
  __shared__ float Xt[40 * 33];
  __shared__ float NHs[64 * 33];     // NHs[j*33 + d] = h0[b,j,d]
  __shared__ float red[4 * 128];

  const int b   = blockIdx.x;
  const int tid = threadIdx.x;
  const float* xb = x + b * 1280;
  for (int idx = tid; idx < 1280; idx += 256) {
    int i = idx >> 5, d = idx & 31;
    Xt[i * 33 + d] = xb[idx];
  }
  const float* nh = NH + b * 2048;
  for (int idx = tid; idx < 2048; idx += 256) {
    int j = idx >> 5, d = idx & 31;
    NHs[j * 33 + d] = nh[idx];
  }
  __syncthreads();

  const int lane = tid & 31;
  const int wid  = tid >> 5;
  const int h    = wid >> 2;
  const int g    = wid & 3;
  const int m    = lane & 15;
  const int half = lane >> 4;
  const int d    = h * 16 + m;

  v8f acc0 = {0.f, 0.f, 0.f, 0.f, 0.f, 0.f, 0.f, 0.f};
  v8f acc1 = {0.f, 0.f, 0.f, 0.f, 0.f, 0.f, 0.f, 0.f};

  const unsigned* B0p = W1B + (unsigned)((2 * g) * 32 + lane) * 8;
  const unsigned* B1p = W1B + (unsigned)((2 * g + 1) * 32 + lane) * 8;

  for (int ks = 0; ks < 80; ++ks) {
    v16bf A;
#pragma unroll
    for (int r = 0; r < 8; ++r) {
      int klo = ks * 32 + a_kbase(r, half);
      int khi = klo + 1;
      int i0 = klo >> 6;  int j0 = klo & 63;
      int i1 = khi >> 6;  int j1 = khi & 63;
      float a0 = Xt[i0 * 33 + d] * NHs[j0 * 33 + d];
      float a1 = Xt[i1 * 33 + d] * NHs[j1 * 33 + d];
      A[2 * r]     = (__bf16)a0;
      A[2 * r + 1] = (__bf16)a1;
    }
    const v16bf Bv0 = *(const v16bf*)(B0p + (unsigned)ks * 2048);
    const v16bf Bv1 = *(const v16bf*)(B1p + (unsigned)ks * 2048);
    acc0 = __builtin_amdgcn_wmma_f32_16x16x32_bf16(false, A, false, Bv0,
                                                   (short)0, acc0, false, false);
    acc1 = __builtin_amdgcn_wmma_f32_16x16x32_bf16(false, A, false, Bv1,
                                                   (short)0, acc1, false, false);
  }

#pragma unroll
  for (int t = 0; t < 2; ++t) {
    v8f acc = t ? acc1 : acc0;
    int s = (2 * g + t) * 16 + m;
    float bias = b1[s];
    float ps = 0.f;
#pragma unroll
    for (int r = 0; r < 8; ++r) {
      float q = acc[r] + bias;
      ps += (q > 0.f ? q : 0.f);
    }
    red[(h * 2 + half) * 128 + s] = ps;
  }
  __syncthreads();
  if (tid < 128) {
    out[b * 192 + 64 + tid] =
        (red[tid] + red[128 + tid]) + (red[256 + tid] + red[384 + tid]);
  }
}

// ---------------------------------------------------------------------------
// Workspace layout (bytes):
//   [0, 409600)            W0B  bf16-swizzled (50*8*32*8 u32)
//   [409600, 1064960)      W1B  bf16-swizzled (80*8*32*8 u32)
//   [1064960, 17842176)    NH   next_hidden f32 [B][64][32]
// ---------------------------------------------------------------------------
extern "C" void kernel_launch(void* const* d_in, const int* in_sizes, int n_in,
                              void* d_out, int out_size, void* d_ws, size_t ws_size,
                              hipStream_t stream) {
  const float* x  = (const float*)d_in[0];
  const float* W0 = (const float*)d_in[1];
  const float* b0 = (const float*)d_in[2];
  const float* W1 = (const float*)d_in[3];
  const float* b1 = (const float*)d_in[4];
  float* out = (float*)d_out;

  unsigned* W0B = (unsigned*)d_ws;
  unsigned* W1B = W0B + 102400;
  float*    NH  = (float*)(W1B + 163840);

  prep_w_kernel<<<400, 256, 0, stream>>>(W0, W0B, 50);
  prep_w_kernel<<<640, 256, 0, stream>>>(W1, W1B, 80);
  layer0_kernel<<<2048, 256, 0, stream>>>(x, W0B, b0, NH, out);
  layer1_kernel<<<2048, 256, 0, stream>>>(x, W1B, b1, NH, out);
}
